// MultiHeadAttention_46875273069091
// MI455X (gfx1250) — matrix operand order repaired and checked by reading, not compiled, not run on hardware
//
#include <hip/hip_runtime.h>
#include <math.h>

// ---------------------------------------------------------------------------
// MI455X (gfx1250) fused MHA block.
// Heavy matmuls: v_wmma_f32_16x16x32_bf16 (bf16 in, f32 accumulate).
// Tile staging: global_load_async_to_lds_b128 + s_wait_asynccnt (CDNA5 async
// path, ASYNCcnt-tracked), double-buffered with one barrier per K-step.
// ---------------------------------------------------------------------------

typedef __attribute__((ext_vector_type(16))) __bf16 bf16x16;
typedef __attribute__((ext_vector_type(8)))  float  v8f;

#define SEQ    4096
#define D_IN   4096
#define D_MODEL 512
#define HEADS  8

#define TILE_M 128
#define TILE_N 128
#define TILE_K 32
#define LDS_STRIDE 40   // 32 + 8 bf16 pad (80B rows) to break bank conflicts

union FragCast {
    struct { uint4 lo; uint4 hi; } u;
    bf16x16 v;
};

// CDNA5 async copy: per-lane 16B  global -> LDS, no VGPR data staging.
// VDST VGPR carries the wave-relative LDS byte offset (low 32 bits of the
// generic shared-memory address); ASYNCcnt tracks completion.
__device__ __forceinline__ void async_copy16(unsigned lds_off, const void* gptr)
{
    asm volatile("global_load_async_to_lds_b128 %0, %1, off"
                 :: "v"(lds_off), "v"(gptr)
                 : "memory");
}

__device__ __forceinline__ void wait_async_all()
{
    asm volatile("s_wait_asynccnt 0" ::: "memory");
}

__device__ __forceinline__ unsigned lds_offset_of(const void* p)
{
    return (unsigned)(size_t)p;   // generic LDS pointer: low 32 bits = offset
}

// ---------------------------------------------------------------------------
// Generic NT GEMM:  C[M,N] = scale * (A[M,K] @ B[N,K]^T) + bias[N]
// OUT_MODE: 0 = bf16 row-major, 1 = f32 row-major, 2 = bf16 transposed (C^T)
// Batched over gridDim.z with explicit strides (A may be shared: strideA=0).
// Block = 256 threads = 8 waves; wave (wm,wn) owns a 64x32 sub-tile
// = 4x2 accumulators of 16x16 f32 (v8f each).
// ---------------------------------------------------------------------------
template <int OUT_MODE>
__global__ __launch_bounds__(256)
void wmma_gemm_nt(const __bf16* __restrict__ A,
                  const __bf16* __restrict__ B,
                  void* __restrict__ Cout,
                  const float* __restrict__ bias,
                  int M, int N, int Kd,
                  int lda, int ldb, int ldc,
                  float scale,
                  long strideA, long strideB, long strideC, long strideBias)
{
    __shared__ __align__(16) __bf16 lsA[2][TILE_M * LDS_STRIDE];
    __shared__ __align__(16) __bf16 lsB[2][TILE_N * LDS_STRIDE];

    const int tid  = threadIdx.x;
    const int lane = tid & 31;
    const int wave = tid >> 5;
    const int bz   = blockIdx.z;

    const __bf16* Ab = A + (long)bz * strideA;
    const __bf16* Bb = B + (long)bz * strideB;

    const int m0 = blockIdx.y * TILE_M;
    const int n0 = blockIdx.x * TILE_N;

    const int wm = wave & 1;   // 0..1 : 64-row group
    const int wn = wave >> 1;  // 0..3 : 32-col group

    v8f acc[4][2];
#pragma unroll
    for (int i = 0; i < 4; ++i)
#pragma unroll
        for (int j = 0; j < 2; ++j) {
            v8f z = {};
            acc[i][j] = z;
        }

    // per-lane fragment indexing (ISA 7.12.2 16-bit layouts, wave32)
    const int mloc  = lane & 15;
    const int khalf = lane >> 4;

    // cooperative tile load mapping: 256 threads x 16 bf16 = 128x32 tile
    const int ldrow = tid >> 1;          // 0..127
    const int ldcol = (tid & 1) * 16;    // 0 or 16

    const unsigned lsA_off[2] = {
        lds_offset_of(&lsA[0][ldrow * LDS_STRIDE + ldcol]),
        lds_offset_of(&lsA[1][ldrow * LDS_STRIDE + ldcol]) };
    const unsigned lsB_off[2] = {
        lds_offset_of(&lsB[0][ldrow * LDS_STRIDE + ldcol]),
        lds_offset_of(&lsB[1][ldrow * LDS_STRIDE + ldcol]) };

    // issue one K-tile (128x32 of A and of B) as async global->LDS copies
    auto issue_tile = [&](int k0, int buf) {
        const __bf16* gA = Ab + (long)(m0 + ldrow) * lda + k0 + ldcol;
        const __bf16* gB = Bb + (long)(n0 + ldrow) * ldb + k0 + ldcol;
        async_copy16(lsA_off[buf],      gA);
        async_copy16(lsA_off[buf] + 16, gA + 8);
        async_copy16(lsB_off[buf],      gB);
        async_copy16(lsB_off[buf] + 16, gB + 8);
    };

    issue_tile(0, 0);
    wait_async_all();
    __syncthreads();

    int buf = 0;
    for (int k0 = 0; k0 < Kd; k0 += TILE_K) {
        if (k0 + TILE_K < Kd)
            issue_tile(k0 + TILE_K, buf ^ 1);

        // B fragments (32x16, KxN): lane n = mloc, lane-half selects K 0-15/16-31
        bf16x16 bf[2];
#pragma unroll
        for (int j = 0; j < 2; ++j) {
            const int n = wn * 32 + j * 16 + mloc;
            const __bf16* p = &lsB[buf][n * LDS_STRIDE + khalf * 16];
            FragCast fc;
            fc.u.lo = *reinterpret_cast<const uint4*>(p);
            fc.u.hi = *reinterpret_cast<const uint4*>(p + 8);
            bf[j] = fc.v;
        }

#pragma unroll
        for (int i = 0; i < 4; ++i) {
            // A fragment (16x32, MxK): lane m = mloc; VGPR0-7 hold K=khalf*16+0..15
            const int m = wm * 64 + i * 16 + mloc;
            const __bf16* p = &lsA[buf][m * LDS_STRIDE + khalf * 16];
            FragCast fa;
            fa.u.lo = *reinterpret_cast<const uint4*>(p);
            fa.u.hi = *reinterpret_cast<const uint4*>(p + 8);
            bf16x16 af = fa.v;
#pragma unroll
            for (int j = 0; j < 2; ++j) {
                acc[i][j] = __builtin_amdgcn_wmma_f32_16x16x32_bf16(
                    false, af, false, bf[j], (short)0, acc[i][j], false, false);
            }
        }

        // own async copies done + every wave finished reading buf -> safe swap
        wait_async_all();
        __syncthreads();
        buf ^= 1;
    }

    // epilogue: C/D layout — lane n = mloc, VGPR r -> row m = r + khalf*8
    const long cbase = (long)bz * strideC;
#pragma unroll
    for (int i = 0; i < 4; ++i) {
        const int mbase = m0 + wm * 64 + i * 16;
#pragma unroll
        for (int j = 0; j < 2; ++j) {
            const int n = n0 + wn * 32 + j * 16 + mloc;
            const float bval = bias ? bias[(long)bz * strideBias + n] : 0.0f;
#pragma unroll
            for (int r = 0; r < 8; ++r) {
                const int row = mbase + r + khalf * 8;
                const float v = acc[i][j][r] * scale + bval;
                if (OUT_MODE == 1) {
                    reinterpret_cast<float*>(Cout)[cbase + (long)row * ldc + n] = v;
                } else if (OUT_MODE == 0) {
                    reinterpret_cast<__bf16*>(Cout)[cbase + (long)row * ldc + n] = (__bf16)v;
                } else { // transposed bf16 store (C^T)
                    reinterpret_cast<__bf16*>(Cout)[cbase + (long)n * ldc + row] = (__bf16)v;
                }
            }
        }
    }
}

// ---------------------------------------------------------------------------
// f32 -> bf16 conversion (grid-stride)
// ---------------------------------------------------------------------------
__global__ __launch_bounds__(256)
void f32_to_bf16(const float* __restrict__ in, __bf16* __restrict__ out, long n)
{
    long i = (long)blockIdx.x * blockDim.x + threadIdx.x;
    const long stride = (long)gridDim.x * blockDim.x;
    for (; i < n; i += stride)
        out[i] = (__bf16)in[i];
}

// ---------------------------------------------------------------------------
// row-wise softmax: S f32 [rows, n] -> P bf16. One 256-thread block per row.
// ---------------------------------------------------------------------------
__global__ __launch_bounds__(256)
void softmax_rows(const float* __restrict__ S, __bf16* __restrict__ P, int n)
{
    const long base = (long)blockIdx.x * n;
    __shared__ float red[8];

    float mx = -3.402823466e38f;
    for (int c = threadIdx.x; c < n; c += 256)
        mx = fmaxf(mx, S[base + c]);
#pragma unroll
    for (int o = 16; o > 0; o >>= 1)
        mx = fmaxf(mx, __shfl_xor(mx, o, 32));
    if ((threadIdx.x & 31) == 0) red[threadIdx.x >> 5] = mx;
    __syncthreads();
    float m_all = red[0];
#pragma unroll
    for (int i = 1; i < 8; ++i) m_all = fmaxf(m_all, red[i]);
    __syncthreads();

    float sum = 0.0f;
    for (int c = threadIdx.x; c < n; c += 256)
        sum += __expf(S[base + c] - m_all);
#pragma unroll
    for (int o = 16; o > 0; o >>= 1)
        sum += __shfl_xor(sum, o, 32);
    if ((threadIdx.x & 31) == 0) red[threadIdx.x >> 5] = sum;
    __syncthreads();
    float tot = 0.0f;
#pragma unroll
    for (int i = 0; i < 8; ++i) tot += red[i];
    const float inv = 1.0f / tot;

    for (int c = threadIdx.x; c < n; c += 256)
        P[base + c] = (__bf16)(__expf(S[base + c] - m_all) * inv);
}

// ---------------------------------------------------------------------------
// residual + LayerNorm, in place on Y (= projection output): one block per row
// Y = LN(Y + X) * gamma + beta
// ---------------------------------------------------------------------------
__global__ __launch_bounds__(256)
void residual_layernorm(float* __restrict__ Y, const float* __restrict__ X,
                        const float* __restrict__ gamma,
                        const float* __restrict__ beta, int n)
{
    const long base = (long)blockIdx.x * n;
    __shared__ float rs[8], rss[8];

    float s = 0.0f, ss = 0.0f;
    for (int c = threadIdx.x; c < n; c += 256) {
        const float y = Y[base + c] + X[base + c];
        s += y;
        ss += y * y;
    }
#pragma unroll
    for (int o = 16; o > 0; o >>= 1) {
        s  += __shfl_xor(s,  o, 32);
        ss += __shfl_xor(ss, o, 32);
    }
    if ((threadIdx.x & 31) == 0) { rs[threadIdx.x >> 5] = s; rss[threadIdx.x >> 5] = ss; }
    __syncthreads();
    s = 0.0f; ss = 0.0f;
#pragma unroll
    for (int i = 0; i < 8; ++i) { s += rs[i]; ss += rss[i]; }
    const float mu  = s / n;
    const float var = ss / n - mu * mu;
    const float inv = rsqrtf(var + 1e-5f);

    for (int c = threadIdx.x; c < n; c += 256) {
        const float y = Y[base + c] + X[base + c];
        Y[base + c] = (y - mu) * inv * gamma[c] + beta[c];
    }
}

// ---------------------------------------------------------------------------
// host-side orchestration
// ---------------------------------------------------------------------------
extern "C" void kernel_launch(void* const* d_in, const int* in_sizes, int n_in,
                              void* d_out, int out_size, void* d_ws, size_t ws_size,
                              hipStream_t stream)
{
    (void)in_sizes; (void)n_in; (void)out_size; (void)ws_size;

    const float* x      = (const float*)d_in[0];
    const float* Pk     = (const float*)d_in[1];
    const float* Pq     = (const float*)d_in[2];
    const float* Pv     = (const float*)d_in[3];
    const float* Wk     = (const float*)d_in[4];
    const float* Wq     = (const float*)d_in[5];
    const float* Wv     = (const float*)d_in[6];
    const float* bk     = (const float*)d_in[7];
    const float* bq     = (const float*)d_in[8];
    const float* bv     = (const float*)d_in[9];
    const float* W_last = (const float*)d_in[10];
    const float* b_last = (const float*)d_in[11];
    const float* gamma  = (const float*)d_in[12];
    const float* beta   = (const float*)d_in[13];
    float* out = (float*)d_out;

    // workspace carve-up (bf16 unless noted)
    size_t off = 0;
    auto carve = [&](size_t bytes) -> void* {
        void* p = (char*)d_ws + off;
        off += (bytes + 255) & ~(size_t)255;
        return p;
    };
    __bf16* xb   = (__bf16*)carve((size_t)SEQ * D_IN * 2);
    __bf16* Pkb  = (__bf16*)carve((size_t)D_MODEL * D_IN * 2);
    __bf16* Pqb  = (__bf16*)carve((size_t)D_MODEL * D_IN * 2);
    __bf16* Pvb  = (__bf16*)carve((size_t)D_MODEL * D_IN * 2);
    __bf16* Wkb  = (__bf16*)carve((size_t)HEADS * D_MODEL * D_MODEL * 2);
    __bf16* Wqb  = (__bf16*)carve((size_t)HEADS * D_MODEL * D_MODEL * 2);
    __bf16* Wvb  = (__bf16*)carve((size_t)HEADS * D_MODEL * D_MODEL * 2);
    __bf16* Wlb  = (__bf16*)carve((size_t)D_IN * D_IN * 2);
    __bf16* Kb   = (__bf16*)carve((size_t)SEQ * D_MODEL * 2);
    __bf16* Qb   = (__bf16*)carve((size_t)SEQ * D_MODEL * 2);
    __bf16* Vb   = (__bf16*)carve((size_t)SEQ * D_MODEL * 2);
    __bf16* Khb  = (__bf16*)carve((size_t)HEADS * SEQ * D_MODEL * 2);
    __bf16* Qhb  = (__bf16*)carve((size_t)HEADS * SEQ * D_MODEL * 2);
    __bf16* VhTb = (__bf16*)carve((size_t)HEADS * D_MODEL * SEQ * 2); // [h][e][t]
    float*  Sbuf = (float*) carve((size_t)SEQ * SEQ * 4);
    __bf16* Pbuf = (__bf16*)carve((size_t)SEQ * SEQ * 2);
    __bf16* Ob   = (__bf16*)carve((size_t)SEQ * D_IN * 2);

    const dim3 blk(256);

    // 0) convert inputs/weights to bf16
    f32_to_bf16<<<4096, blk, 0, stream>>>(x,      xb,  (long)SEQ * D_IN);
    f32_to_bf16<<<2048, blk, 0, stream>>>(Pk,     Pkb, (long)D_MODEL * D_IN);
    f32_to_bf16<<<2048, blk, 0, stream>>>(Pq,     Pqb, (long)D_MODEL * D_IN);
    f32_to_bf16<<<2048, blk, 0, stream>>>(Pv,     Pvb, (long)D_MODEL * D_IN);
    f32_to_bf16<<<2048, blk, 0, stream>>>(Wk,     Wkb, (long)HEADS * D_MODEL * D_MODEL);
    f32_to_bf16<<<2048, blk, 0, stream>>>(Wq,     Wqb, (long)HEADS * D_MODEL * D_MODEL);
    f32_to_bf16<<<2048, blk, 0, stream>>>(Wv,     Wvb, (long)HEADS * D_MODEL * D_MODEL);
    f32_to_bf16<<<4096, blk, 0, stream>>>(W_last, Wlb, (long)D_IN * D_IN);

    // 1) shared projections: [4096,4096] @ [512,4096]^T -> [4096,512] bf16
    {
        dim3 grid(D_MODEL / TILE_N, SEQ / TILE_M, 1);
        wmma_gemm_nt<0><<<grid, blk, 0, stream>>>(xb, Pkb, Kb, nullptr,
            SEQ, D_MODEL, D_IN, D_IN, D_IN, D_MODEL, 1.0f, 0, 0, 0, 0);
        wmma_gemm_nt<0><<<grid, blk, 0, stream>>>(xb, Pqb, Qb, nullptr,
            SEQ, D_MODEL, D_IN, D_IN, D_IN, D_MODEL, 1.0f, 0, 0, 0, 0);
        wmma_gemm_nt<0><<<grid, blk, 0, stream>>>(xb, Pvb, Vb, nullptr,
            SEQ, D_MODEL, D_IN, D_IN, D_IN, D_MODEL, 1.0f, 0, 0, 0, 0);
    }

    // 2) per-head linears (batched over gridDim.z): Xh = X @ W[h]^T + b[h]
    {
        dim3 grid(D_MODEL / TILE_N, SEQ / TILE_M, HEADS);
        const long wstr = (long)D_MODEL * D_MODEL;
        const long ostr = (long)SEQ * D_MODEL;
        wmma_gemm_nt<0><<<grid, blk, 0, stream>>>(Kb, Wkb, Khb, bk,
            SEQ, D_MODEL, D_MODEL, D_MODEL, D_MODEL, D_MODEL, 1.0f,
            0, wstr, ostr, D_MODEL);
        wmma_gemm_nt<0><<<grid, blk, 0, stream>>>(Qb, Wqb, Qhb, bq,
            SEQ, D_MODEL, D_MODEL, D_MODEL, D_MODEL, D_MODEL, 1.0f,
            0, wstr, ostr, D_MODEL);
        // V stored transposed ([e][t]) so PV is also an NT GEMM
        wmma_gemm_nt<2><<<grid, blk, 0, stream>>>(Vb, Wvb, VhTb, bv,
            SEQ, D_MODEL, D_MODEL, D_MODEL, D_MODEL, SEQ, 1.0f,
            0, wstr, (long)D_MODEL * SEQ, D_MODEL);
    }

    // 3) attention per head: S = scale*Qh@Kh^T ; P = softmax(S) ; O = P@Vh
    const float scale = 1.0f / sqrtf((float)D_MODEL);
    for (int h = 0; h < HEADS; ++h) {
        const __bf16* Qh  = Qhb  + (long)h * SEQ * D_MODEL;
        const __bf16* Kh  = Khb  + (long)h * SEQ * D_MODEL;
        const __bf16* VhT = VhTb + (long)h * D_MODEL * SEQ;

        dim3 gridS(SEQ / TILE_N, SEQ / TILE_M, 1);
        wmma_gemm_nt<1><<<gridS, blk, 0, stream>>>(Qh, Kh, Sbuf, nullptr,
            SEQ, SEQ, D_MODEL, D_MODEL, D_MODEL, SEQ, scale, 0, 0, 0, 0);

        softmax_rows<<<SEQ, blk, 0, stream>>>(Sbuf, Pbuf, SEQ);

        dim3 gridO(D_MODEL / TILE_N, SEQ / TILE_M, 1);
        wmma_gemm_nt<0><<<gridO, blk, 0, stream>>>(Pbuf, VhT, Ob + (long)h * D_MODEL,
            nullptr, SEQ, D_MODEL, SEQ, SEQ, SEQ, D_IN, 1.0f, 0, 0, 0, 0);
    }

    // 4) output projection -> f32 directly into d_out
    {
        dim3 grid(D_IN / TILE_N, SEQ / TILE_M, 1);
        wmma_gemm_nt<1><<<grid, blk, 0, stream>>>(Ob, Wlb, out, b_last,
            SEQ, D_IN, D_IN, D_IN, D_IN, D_IN, 1.0f, 0, 0, 0, 0);
    }

    // 5) residual + LayerNorm (in place on d_out)
    residual_layernorm<<<SEQ, blk, 0, stream>>>(out, x, gamma, beta, D_IN);
}
